// GraphAttentionLayer_59115929862516
// MI455X (gfx1250) — matrix operand order repaired and checked by reading, not compiled, not run on hardware
//
#include <hip/hip_runtime.h>

#define ALPHA   0.2f
#define NEG_INF (-9.0e15f)
#define Bb   8
#define Nn   1024
#define Ff   512
#define Hh   8
#define DH   64

typedef __attribute__((ext_vector_type(16))) __bf16 v16bf;
typedef __attribute__((ext_vector_type(8)))  float  v8f;
typedef unsigned short u16;
typedef unsigned int   u32;

union Frag32 {          // 32 bytes: one bf16 WMMA A/B fragment
    u32   w[8];
    uint4 u[2];
    v16bf v;
};

static __device__ __forceinline__ u16 f2bf(float f) {
    u32 u = __float_as_uint(f);
    u32 r = u + 0x7FFFu + ((u >> 16) & 1u);   // round-to-nearest-even
    return (u16)(r >> 16);
}
static __device__ __forceinline__ float bf2f(u16 h) {
    return __uint_as_float(((u32)h) << 16);
}
static __device__ __forceinline__ v8f wmma_bf16(v16bf a, v16bf b, v8f c) {
    return __builtin_amdgcn_wmma_f32_16x16x32_bf16(false, a, false, b, (short)0, c, false, false);
}

// ---------------- split f32 -> bf16 hi + bf16 lo ----------------
__global__ void k_split_bf16(const float* __restrict__ src,
                             u16* __restrict__ hi, u16* __restrict__ lo, int n) {
    int i = blockIdx.x * blockDim.x + threadIdx.x;
    if (i >= n) return;
    float f = src[i];
    u16 h = f2bf(f);
    hi[i] = h;
    lo[i] = f2bf(f - bf2f(h));
}

// ---------------- pack adjacency into bitmasks ----------------
// one wave per (b, i) row; 32 chunks of 32 columns
__global__ __launch_bounds__(32) void k_pack_adj(const int* __restrict__ adj,
                                                 u32* __restrict__ adjp) {
    int row = blockIdx.x;                 // b*N + i, [0, 8192)
    int l = threadIdx.x;
    const int* base = adj + (size_t)row * Nn;
    u32* op = adjp + (size_t)row * 32;
    for (int c = 0; c < 32; ++c) {
        int pred = base[c * 32 + l] > 0;
        unsigned long long m = __ballot(pred);
        if (l == 0) op[c] = (u32)m;
    }
}

// ---------------- h = x @ W + b  (split-bf16, WMMA) ----------------
// one wave per 16x64 output tile; grid = (rows/16, Fout/64)
// A fragment reused across 4 column tiles -> 12 WMMA per 4 b128 + 128 u16 loads
__global__ __launch_bounds__(32) void k_gemm(const u16* __restrict__ xh, const u16* __restrict__ xl,
                                             const u16* __restrict__ wh, const u16* __restrict__ wl,
                                             const float* __restrict__ bias,
                                             float* __restrict__ h) {
    int row0 = blockIdx.x * 16, col0 = blockIdx.y * 64;
    int l = threadIdx.x, r = l & 15, g = l >> 4;
    v8f acc[4];
    #pragma unroll
    for (int t = 0; t < 4; ++t)
        #pragma unroll
        for (int i = 0; i < 8; ++i) acc[t][i] = 0.f;

    size_t abase = (size_t)(row0 + r) * Ff;
    #pragma unroll 2
    for (int kc = 0; kc < Ff / 32; ++kc) {
        int K0 = kc * 32;
        Frag32 ah, al;
        // A fragment: row r, K = [K0+8g, +8) and [K0+16+8g, +8) (contiguous)
        ah.u[0] = *(const uint4*)(xh + abase + K0 + 8 * g);
        ah.u[1] = *(const uint4*)(xh + abase + K0 + 16 + 8 * g);
        al.u[0] = *(const uint4*)(xl + abase + K0 + 8 * g);
        al.u[1] = *(const uint4*)(xl + abase + K0 + 16 + 8 * g);
        // B fragments: column r, K = [K0+8g, +8) and [K0+16+8g, +8), 4 column tiles of 16 at col0
        size_t bbase = (size_t)K0 * Ff + col0 + r;
        #pragma unroll
        for (int t = 0; t < 4; ++t) {
            Frag32 bh, bl;
            #pragma unroll
            for (int w = 0; w < 8; ++w) {
                int k0 = (w < 4) ? (8 * g + 2 * w) : (8 + 8 * g + 2 * w);
                size_t o0 = bbase + (size_t)k0 * Ff + 16 * t, o1 = o0 + Ff;
                bh.w[w] = (u32)wh[o0] | ((u32)wh[o1] << 16);
                bl.w[w] = (u32)wl[o0] | ((u32)wl[o1] << 16);
            }
            acc[t] = wmma_bf16(ah.v, bh.v, acc[t]);
            acc[t] = wmma_bf16(ah.v, bl.v, acc[t]);
            acc[t] = wmma_bf16(al.v, bh.v, acc[t]);
        }
    }
    #pragma unroll
    for (int t = 0; t < 4; ++t) {
        float bv = bias[col0 + 16 * t + r];
        #pragma unroll
        for (int v = 0; v < 8; ++v) {
            int m = v + 8 * g;
            h[(size_t)(row0 + m) * Ff + col0 + 16 * t + r] = acc[t][v] + bv;
        }
    }
}

// ---------------- prep: h -> bf16 hi/lo, per-head s_src/s_dst ----------------
__global__ void k_prep(const float* __restrict__ h, const float* __restrict__ att,
                       u16* __restrict__ hhi, u16* __restrict__ hlo,
                       float* __restrict__ ssrc, float* __restrict__ sdst) {
    int t = blockIdx.x * blockDim.x + threadIdx.x;     // 65536 = 8192 rows * 8 heads
    int head = t & 7;
    int row  = t >> 3;
    size_t base = (size_t)row * Ff + head * DH;
    const float4* p4 = (const float4*)(h + base);
    const float* asrc = att + head * 2 * DH;
    const float* adst = asrc + DH;
    float ss = 0.f, sd = 0.f;
    #pragma unroll
    for (int q = 0; q < DH / 4; ++q) {
        float4 v = p4[q];
        int d = q * 4;
        ss += v.x * asrc[d] + v.y * asrc[d + 1] + v.z * asrc[d + 2] + v.w * asrc[d + 3];
        sd += v.x * adst[d] + v.y * adst[d + 1] + v.z * adst[d + 2] + v.w * adst[d + 3];
        float f[4] = {v.x, v.y, v.z, v.w};
        #pragma unroll
        for (int j = 0; j < 4; ++j) {
            u16 hb = f2bf(f[j]);
            hhi[base + d + j] = hb;
            hlo[base + d + j] = f2bf(f[j] - bf2f(hb));
        }
    }
    int b = row >> 10, n = row & (Nn - 1);
    size_t sidx = ((size_t)b * Hh + head) * Nn + n;
    ssrc[sidx] = ss;
    sdst[sidx] = sd;
}

// ---------------- flash-GAT: softmax(P) @ h per head, ELU ----------------
// one wave per (b, head, 16-row i-tile); grid = B*H*(N/16) = 4096
__global__ __launch_bounds__(32) void k_gat(const u16* __restrict__ hhi, const u16* __restrict__ hlo,
                                            const float* __restrict__ ssrc, const float* __restrict__ sdst,
                                            const u32* __restrict__ adjp,
                                            float* __restrict__ out) {
    int bid = blockIdx.x;
    int it = bid & 63, head = (bid >> 6) & 7, b = bid >> 9;
    int l = threadIdx.x, r = l & 15, g = l >> 4;
    int i0 = it * 16;

    const float* ssr = ssrc + ((size_t)b * Hh + head) * Nn;
    float sd = sdst[((size_t)b * Hh + head) * Nn + i0 + r];

    float m = NEG_INF, S = 0.f;
    v8f acc[4];
    #pragma unroll
    for (int t = 0; t < 4; ++t)
        #pragma unroll
        for (int i = 0; i < 8; ++i) acc[t][i] = 0.f;

    for (int jc = 0; jc < Nn / 32; ++jc) {
        int j0 = jc * 32;
        float myss = ssr[j0 + l];
        u32 aw = adjp[(size_t)(b * Nn + i0 + r) * 32 + jc];

        // scores for exactly the 16 K-slots this lane feeds into the A fragment
        float p[16];
        float cmax = NEG_INF;
        #pragma unroll
        for (int e = 0; e < 16; ++e) {
            int kk = (e < 8) ? (e + 8 * g) : (8 + e + 8 * g);   // K within chunk
            float sv = sd + __shfl(myss, kk, 32);
            sv = sv > 0.f ? sv : ALPHA * sv;                     // leaky relu
            sv = ((aw >> kk) & 1u) ? sv : NEG_INF;               // mask (finite, like ref)
            p[e] = sv;
            cmax = fmaxf(cmax, sv);
        }
        cmax = fmaxf(cmax, __shfl_xor(cmax, 16, 32));            // combine row halves
        float newm = fmaxf(m, cmax);
        float scale = __expf(m - newm);
        m = newm;

        float csum = 0.f;
        #pragma unroll
        for (int e = 0; e < 16; ++e) { p[e] = __expf(p[e] - newm); csum += p[e]; }
        S = S * scale + csum;

        // rescale accumulators by per-row scale (lane 'row' holds it)
        #pragma unroll
        for (int v = 0; v < 8; ++v) {
            float sc = __shfl(scale, v + 8 * g, 32);
            #pragma unroll
            for (int t = 0; t < 4; ++t) acc[t][v] *= sc;
        }

        // P fragment, split hi/lo bf16
        Frag32 phi, plo;
        #pragma unroll
        for (int w = 0; w < 8; ++w) {
            float p0 = p[2 * w], p1 = p[2 * w + 1];
            u16 h0 = f2bf(p0), h1 = f2bf(p1);
            phi.w[w] = (u32)h0 | ((u32)h1 << 16);
            plo.w[w] = (u32)f2bf(p0 - bf2f(h0)) | ((u32)f2bf(p1 - bf2f(h1)) << 16);
        }

        // B fragments: lane = d-column r, K-rows j0 + table order, all 64 d-columns of this head
        size_t hbase = ((size_t)(b * Nn) + j0) * Ff + head * DH + r;
        if (jc + 1 < Nn / 32) {
            __builtin_prefetch((const void*)(hhi + hbase + 32 * Ff), 0, 1);
            __builtin_prefetch((const void*)(hlo + hbase + 32 * Ff), 0, 1);
        }
        #pragma unroll
        for (int t = 0; t < 4; ++t) {
            Frag32 fh, fl;
            #pragma unroll
            for (int w = 0; w < 8; ++w) {
                int k0 = (w < 4) ? (8 * g + 2 * w) : (8 + 8 * g + 2 * w);
                size_t o0 = hbase + (size_t)k0 * Ff + 16 * t, o1 = o0 + Ff;
                fh.w[w] = (u32)hhi[o0] | ((u32)hhi[o1] << 16);
                fl.w[w] = (u32)hlo[o0] | ((u32)hlo[o1] << 16);
            }
            v16bf bh = fh.v, bl = fl.v;
            acc[t] = wmma_bf16(phi.v, bh, acc[t]);
            acc[t] = wmma_bf16(phi.v, bl, acc[t]);
            acc[t] = wmma_bf16(plo.v, bh, acc[t]);
        }
    }

    S = S + __shfl_xor(S, 16, 32);
    float inv = 1.f / S;
    #pragma unroll
    for (int v = 0; v < 8; ++v) {
        int mrow = v + 8 * g;
        float iv = __shfl(inv, mrow, 32);
        #pragma unroll
        for (int t = 0; t < 4; ++t) {
            float val = acc[t][v] * iv;
            val = val > 0.f ? val : expm1f(val);                  // ELU(alpha=1)
            out[((size_t)(b * Nn) + i0 + mrow) * Ff + head * DH + t * 16 + r] = val;
        }
    }
}

extern "C" void kernel_launch(void* const* d_in, const int* in_sizes, int n_in,
                              void* d_out, int out_size, void* d_ws, size_t ws_size,
                              hipStream_t stream) {
    const float* x   = (const float*)d_in[0];   // (8,1024,512)
    const int*   adj = (const int*)d_in[1];     // (8,1024,1024)
    const float* Ww  = (const float*)d_in[2];   // (512,512)
    const float* Wb  = (const float*)d_in[3];   // (512,)
    const float* att = (const float*)d_in[4];   // (8,128)
    float* out = (float*)d_out;

    char* ws = (char*)d_ws;
    size_t o = 0;
    float* h    = (float*)(ws + o); o += (size_t)Bb * Nn * Ff * 4;      // 16 MB
    u16*   xh   = (u16*)  (ws + o); o += (size_t)Bb * Nn * Ff * 2;      //  8 MB
    u16*   xl   = (u16*)  (ws + o); o += (size_t)Bb * Nn * Ff * 2;
    u16*   wh   = (u16*)  (ws + o); o += (size_t)Ff * Ff * 2;
    u16*   wl   = (u16*)  (ws + o); o += (size_t)Ff * Ff * 2;
    u16*   hhi  = (u16*)  (ws + o); o += (size_t)Bb * Nn * Ff * 2;
    u16*   hlo  = (u16*)  (ws + o); o += (size_t)Bb * Nn * Ff * 2;
    float* ssv  = (float*)(ws + o); o += (size_t)Bb * Hh * Nn * 4;
    float* sdv  = (float*)(ws + o); o += (size_t)Bb * Hh * Nn * 4;
    u32*   adjp = (u32*)  (ws + o); o += (size_t)Bb * Nn * 32 * 4;

    int nx = Bb * Nn * Ff;                      // 4194304
    int nw = Ff * Ff;                           // 262144
    k_split_bf16<<<(nx + 255) / 256, 256, 0, stream>>>(x,  xh, xl, nx);
    k_split_bf16<<<(nw + 255) / 256, 256, 0, stream>>>(Ww, wh, wl, nw);
    k_pack_adj<<<Bb * Nn, 32, 0, stream>>>(adj, adjp);
    k_gemm<<<dim3(Bb * Nn / 16, Ff / 64), 32, 0, stream>>>(xh, xl, wh, wl, Wb, h);
    k_prep<<<(Bb * Nn * Hh) / 256, 256, 0, stream>>>(h, att, hhi, hlo, ssv, sdv);
    k_gat<<<Bb * Hh * (Nn / 16), 32, 0, stream>>>(hhi, hlo, ssv, sdv, adjp, out);
}
